// Attention_3066606649616
// MI455X (gfx1250) — matrix operand order repaired and checked by reading, not compiled, not run on hardware
//
#include <hip/hip_runtime.h>
#include <hip/hip_bf16.h>

typedef __bf16 bf16_t;
typedef bf16_t v16bf __attribute__((ext_vector_type(16)));
typedef float  v8f   __attribute__((ext_vector_type(8)));

#define B_SZ   2048
#define T_ATT  32
#define D_INP  512
#define HDIM   256
#define NCLS   38
#define NSTEP  26
#define GATES  1024     // 4*H
#define KCOMB  768      // D_INP + HDIM
#define LD_IH  550      // D_INP + NCLS

// ---------- WMMA helpers (CDNA5 gfx1250, wave32) ----------

__device__ inline v8f wmma_bf16(v16bf a, v16bf b, v8f c) {
    return __builtin_amdgcn_wmma_f32_16x16x32_bf16(
        false, a, false, b, (short)0, c, false, false);
}

// A fragment (16x32 bf16) gathered from a row-major bf16 buffer (LDS).
// Layout: lane<16 -> row M=lane, K in {0..15}; lane>=16 -> row
// M=lane-16, K in {16..31}; element pairs per VGPR (matches B swizzle).
__device__ inline v16bf load_a_frag(const bf16_t* base, int ld) {
    int lane = threadIdx.x & 31;
    int m  = lane & 15;
    int kh = (lane >> 4) * 16;
    const bf16_t* row = base + m * ld;
    v16bf a;
#pragma unroll
    for (int v = 0; v < 8; ++v) {
        int k = kh + 2 * v;
        a[2 * v]     = row[k];
        a[2 * v + 1] = row[k + 1];
    }
    return a;
}

// B fragment, pre-swizzled in global memory: 32 lanes x 16 bf16, lane-major.
__device__ inline v16bf load_b_frag(const bf16_t* frag) {
    int lane = threadIdx.x & 31;
    return *(const v16bf*)(frag + lane * 16);   // aligned 32B vector load
}

// D tile (16x16 f32) store: VGPR v -> M = v + (lane>=16 ? 8 : 0), N = lane&15.
__device__ inline void store_d_bf16(bf16_t* dst, int ld, v8f d) {
    int lane = threadIdx.x & 31;
    int n  = lane & 15;
    int mh = (lane >> 4) * 8;
#pragma unroll
    for (int v = 0; v < 8; ++v) dst[(mh + v) * ld + n] = (bf16_t)d[v];
}

__device__ inline v8f zero8() { v8f z = {0.f,0.f,0.f,0.f,0.f,0.f,0.f,0.f}; return z; }

// ---------- prep kernels: swizzle weights into B-fragment order ----------

// Generic: src is [Nsrc][ldsrc] row-major f32; logical B[k][n] = src[n][k].
// Output fragment id = kt*Nt + nt; within a fragment, lane-major 16 elems:
//   lane<16: K = kt*32 + i; lane>=16: K = kt*32 + 16 + i; N = nt*16 + (lane&15)
__global__ void swizzleB_kernel(const float* __restrict__ src, int ldsrc,
                                int Ksrc, int Nsrc, int Kt, int Nt,
                                bf16_t* __restrict__ out) {
    int tid = blockIdx.x * blockDim.x + threadIdx.x;
    int total = Kt * Nt * 32;
    if (tid >= total) return;
    int lane = tid & 31;
    int fid  = tid >> 5;
    int kt = fid / Nt, nt = fid - kt * Nt;
    int n = nt * 16 + (lane & 15);
    int kbase = kt * 32 + ((lane >> 4) ? 16 : 0);
    bf16_t* dst = out + (size_t)fid * 512 + lane * 16;
    for (int i = 0; i < 16; ++i) {
        int k = kbase + i;
        float v = (k < Ksrc && n < Nsrc) ? src[(size_t)n * ldsrc + k] : 0.0f;
        dst[i] = (bf16_t)v;
    }
}

// Combined [W_ih[:, :512] | W_hh] -> K=768, N=1024 fragment array (Kt=24, Nt=64)
__global__ void wg_swizzle_kernel(const float* __restrict__ W_ih,
                                  const float* __restrict__ W_hh,
                                  bf16_t* __restrict__ out) {
    int tid = blockIdx.x * blockDim.x + threadIdx.x;
    if (tid >= 24 * 64 * 32) return;
    int lane = tid & 31;
    int fid  = tid >> 5;
    int kt = fid / 64, nt = fid - kt * 64;
    int n = nt * 16 + (lane & 15);
    int kbase = kt * 32 + ((lane >> 4) ? 16 : 0);
    bf16_t* dst = out + (size_t)fid * 512 + lane * 16;
    for (int i = 0; i < 16; ++i) {
        int k = kbase + i;
        float v = (k < D_INP) ? W_ih[(size_t)n * LD_IH + k]
                              : W_hh[(size_t)n * HDIM + (k - D_INP)];
        dst[i] = (bf16_t)v;
    }
}

// One-hot weight columns (transposed for coalesced reads) + fused gate bias.
__global__ void misc_prep_kernel(const float* __restrict__ W_ih,
                                 const float* __restrict__ b_ih,
                                 const float* __restrict__ b_hh,
                                 float* __restrict__ Wonehot,
                                 float* __restrict__ bcomb) {
    int tid = blockIdx.x * blockDim.x + threadIdx.x;
    if (tid < NCLS * GATES) {
        int ch = tid / GATES, n = tid - ch * GATES;
        Wonehot[ch * GATES + n] = W_ih[(size_t)n * LD_IH + D_INP + ch];
    } else if (tid < NCLS * GATES + GATES) {
        int n = tid - NCLS * GATES;
        bcomb[n] = b_ih[n] + b_hh[n];
    }
}

// ---------- H_proj = batch_H @ W_i2h^T  (M=B*T=65536, N=256, K=512) ----------

__global__ void __launch_bounds__(256)
hproj_kernel(const float* __restrict__ batch_H,
             const bf16_t* __restrict__ Wi2h_frag,
             bf16_t* __restrict__ Hproj) {
    __shared__ bf16_t a_tile[16 * 512];      // 16 KB
    int t = threadIdx.x;
    size_t base = (size_t)blockIdx.x * 16 * 512;
    for (int idx = t; idx < 16 * 512; idx += 256)
        a_tile[idx] = (bf16_t)batch_H[base + idx];
    __syncthreads();

    int w = t >> 5;
    int nt0 = 2 * w, nt1 = nt0 + 1;
    v8f acc0 = zero8(), acc1 = zero8();
    for (int kt = 0; kt < 16; ++kt) {
        v16bf a  = load_a_frag(a_tile + kt * 32, 512);
        v16bf b0 = load_b_frag(Wi2h_frag + (size_t)(kt * 16 + nt0) * 512);
        v16bf b1 = load_b_frag(Wi2h_frag + (size_t)(kt * 16 + nt1) * 512);
        acc0 = wmma_bf16(a, b0, acc0);
        acc1 = wmma_bf16(a, b1, acc1);
    }
    bf16_t* outb = Hproj + (size_t)blockIdx.x * 16 * 256;
    store_d_bf16(outb + nt0 * 16, 256, acc0);
    store_d_bf16(outb + nt1 * 16, 256, acc1);
}

// ---------- persistent decoder: 128 blocks x 16 batch rows, 26 steps ----------

__global__ void __launch_bounds__(256)
decoder_kernel(const float* __restrict__ batch_H,
               const long long* __restrict__ text,     // int64 [B][26]
               const bf16_t* __restrict__ Hproj,       // [B*32][256] bf16
               const float* __restrict__ b_h2h,
               const float* __restrict__ W_score,      // [256]
               const bf16_t* __restrict__ Wh2h_frag,   // Kt=8,  Nt=16
               const bf16_t* __restrict__ Wg_frag,     // Kt=24, Nt=64
               const bf16_t* __restrict__ Wgen_frag,   // Kt=8,  Nt=3
               const float* __restrict__ Wonehot,      // [38][1024]
               const float* __restrict__ bcomb,        // [1024]
               const float* __restrict__ b_gen,        // [38]
               float* __restrict__ out) {              // [B][26][38]
    __shared__ bf16_t xbuf[16 * KCOMB];   // 24 KB: cols 0..511 ctx, 512..767 h
    __shared__ bf16_t gbuf[16 * GATES];   // 32 KB: hp (ld=256) then gates (ld=1024)
    __shared__ float  abuf[16 * T_ATT];   // 2 KB : attention scores/weights

    int t = threadIdx.x, w = t >> 5, lane = t & 31;
    int r0 = blockIdx.x * 16;
    int pm = t >> 4, pj = t & 15;         // pointwise (row, sub-col) ownership

    for (int idx = t; idx < 16 * KCOMB; idx += 256) xbuf[idx] = (bf16_t)0.0f;
    float c_reg[16];
#pragma unroll
    for (int k = 0; k < 16; ++k) c_reg[k] = 0.0f;
    __syncthreads();

    for (int s = 0; s < NSTEP; ++s) {
        // -- A: hp = h @ W_h2h^T  (M=16,N=256,K=256; 2 N-tiles per wave)
        {
            int nt0 = 2 * w, nt1 = nt0 + 1;
            v8f a0 = zero8(), a1 = zero8();
            for (int kt = 0; kt < 8; ++kt) {
                v16bf af = load_a_frag(xbuf + 512 + kt * 32, KCOMB);
                v16bf b0 = load_b_frag(Wh2h_frag + (size_t)(kt * 16 + nt0) * 512);
                v16bf b1 = load_b_frag(Wh2h_frag + (size_t)(kt * 16 + nt1) * 512);
                a0 = wmma_bf16(af, b0, a0);
                a1 = wmma_bf16(af, b1, a1);
            }
            store_d_bf16(gbuf + nt0 * 16, 256, a0);
            store_d_bf16(gbuf + nt1 * 16, 256, a1);
        }
        __syncthreads();

        // -- B1: e[m][tt] = sum_h tanh(Hproj + hp + b_h2h) * W_score  (VALU)
        for (int p = 0; p < 2; ++p) {
            int idx = t * 2 + p;
            int m = idx >> 5, tt = idx & 31;
            const bf16_t* hp  = gbuf + m * 256;
            const bf16_t* hpj = Hproj + ((size_t)(r0 + m) * T_ATT + tt) * 256;
            float acc = 0.f;
            for (int hh = 0; hh < HDIM; ++hh) {
                float v = (float)hpj[hh] + (float)hp[hh] + b_h2h[hh];
                acc += tanhf(v) * W_score[hh];
            }
            abuf[m * T_ATT + tt] = acc;
        }
        __syncthreads();

        // -- B2: softmax over T=32 (one thread per row)
        if (t < 16) {
            float mx = -1e30f;
            for (int i = 0; i < T_ATT; ++i) mx = fmaxf(mx, abuf[t * T_ATT + i]);
            float sm = 0.f;
            for (int i = 0; i < T_ATT; ++i) {
                float e = __expf(abuf[t * T_ATT + i] - mx);
                abuf[t * T_ATT + i] = e; sm += e;
            }
            float inv = 1.0f / sm;
            for (int i = 0; i < T_ATT; ++i) abuf[t * T_ATT + i] *= inv;
        }
        __syncthreads();

        // -- B3: context[m][d] = sum_tt alpha * batch_H (per-row matvec, VALU)
        {
            int d0 = pj * 32;
            float acc[32];
#pragma unroll
            for (int d = 0; d < 32; ++d) acc[d] = 0.f;
            for (int tt = 0; tt < T_ATT; ++tt) {
                float a = abuf[pm * T_ATT + tt];
                const float* bh = batch_H + ((size_t)(r0 + pm) * T_ATT + tt) * D_INP + d0;
#pragma unroll
                for (int d = 0; d < 32; ++d) acc[d] += a * bh[d];
            }
            bf16_t* xr = xbuf + pm * KCOMB + d0;
#pragma unroll
            for (int d = 0; d < 32; ++d) xr[d] = (bf16_t)acc[d];
        }
        __syncthreads();

        // -- C: gates = [context, h] @ [W_ih[:, :512] | W_hh]^T
        //        (M=16, N=1024, K=768; 8 N-tiles per wave)
        {
            v8f acc[8];
#pragma unroll
            for (int i = 0; i < 8; ++i) acc[i] = zero8();
            for (int kt = 0; kt < 24; ++kt) {
                v16bf af = load_a_frag(xbuf + kt * 32, KCOMB);
#pragma unroll
                for (int i = 0; i < 8; ++i) {
                    int nt = w * 8 + i;
                    v16bf bfg = load_b_frag(Wg_frag + (size_t)(kt * 64 + nt) * 512);
                    acc[i] = wmma_bf16(af, bfg, acc[i]);
                }
            }
#pragma unroll
            for (int i = 0; i < 8; ++i)
                store_d_bf16(gbuf + (w * 8 + i) * 16, GATES, acc[i]);
        }
        __syncthreads();

        // -- D: LSTM pointwise (c in registers, one-hot column + bias fused)
        {
            int ch = (int)text[(size_t)(r0 + pm) * NSTEP + s];
            const float* oh = Wonehot + ch * GATES;
            const bf16_t* gr = gbuf + pm * GATES;
#pragma unroll
            for (int k = 0; k < 16; ++k) {
                int j = pj + 16 * k;
                float gi = (float)gr[j]       + bcomb[j]       + oh[j];
                float gf = (float)gr[256 + j] + bcomb[256 + j] + oh[256 + j];
                float gg = (float)gr[512 + j] + bcomb[512 + j] + oh[512 + j];
                float go = (float)gr[768 + j] + bcomb[768 + j] + oh[768 + j];
                float si = 1.f / (1.f + __expf(-gi));
                float sf = 1.f / (1.f + __expf(-gf));
                float so = 1.f / (1.f + __expf(-go));
                float cn = sf * c_reg[k] + si * tanhf(gg);
                c_reg[k] = cn;
                xbuf[pm * KCOMB + 512 + j] = (bf16_t)(so * tanhf(cn));
            }
        }
        __syncthreads();

        // -- E: probs = h_new @ W_gen^T + b_gen  (M=16, N=48 padded, K=256)
        if (w < 3) {
            v8f acc = zero8();
            for (int kt = 0; kt < 8; ++kt) {
                v16bf af  = load_a_frag(xbuf + 512 + kt * 32, KCOMB);
                v16bf bfg = load_b_frag(Wgen_frag + (size_t)(kt * 3 + w) * 512);
                acc = wmma_bf16(af, bfg, acc);
            }
            int n = lane & 15, mh = (lane >> 4) * 8;
            int cidx = w * 16 + n;
            if (cidx < NCLS) {
                float bg = b_gen[cidx];
#pragma unroll
                for (int v = 0; v < 8; ++v) {
                    int m = mh + v;
                    out[((size_t)(r0 + m) * NSTEP + s) * NCLS + cidx] = acc[v] + bg;
                }
            }
        }
        __syncthreads();
    }
}

// ---------- launcher ----------

extern "C" void kernel_launch(void* const* d_in, const int* in_sizes, int n_in,
                              void* d_out, int out_size, void* d_ws, size_t ws_size,
                              hipStream_t stream) {
    const float*     batch_H = (const float*)d_in[0];
    const long long* text    = (const long long*)d_in[1];
    const float*     W_i2h   = (const float*)d_in[2];
    const float*     W_h2h   = (const float*)d_in[3];
    const float*     b_h2h   = (const float*)d_in[4];
    const float*     W_score = (const float*)d_in[5];
    const float*     W_ih    = (const float*)d_in[6];
    const float*     b_ih    = (const float*)d_in[7];
    const float*     W_hh    = (const float*)d_in[8];
    const float*     b_hh    = (const float*)d_in[9];
    const float*     W_gen   = (const float*)d_in[10];
    const float*     b_gen   = (const float*)d_in[11];
    float* out = (float*)d_out;

    char* ws = (char*)d_ws;
    size_t off = 0;
    auto carve = [&](size_t bytes) {
        char* p = ws + off;
        off += (bytes + 255) & ~(size_t)255;
        return p;
    };
    bf16_t* Wg_frag   = (bf16_t*)carve((size_t)24 * 64 * 512 * 2);   // 1.5 MB
    bf16_t* Wh2h_frag = (bf16_t*)carve((size_t)8 * 16 * 512 * 2);
    bf16_t* Wgen_frag = (bf16_t*)carve((size_t)8 * 3 * 512 * 2);
    bf16_t* Wi2h_frag = (bf16_t*)carve((size_t)16 * 16 * 512 * 2);
    float*  Wonehot   = (float*)carve((size_t)NCLS * GATES * 4);
    float*  bcomb     = (float*)carve((size_t)GATES * 4);
    bf16_t* Hproj     = (bf16_t*)carve((size_t)B_SZ * T_ATT * HDIM * 2); // 33.5 MB

    swizzleB_kernel<<<(8 * 16 * 32 + 255) / 256, 256, 0, stream>>>(
        W_h2h, 256, 256, 256, 8, 16, Wh2h_frag);
    swizzleB_kernel<<<(8 * 3 * 32 + 255) / 256, 256, 0, stream>>>(
        W_gen, 256, 256, NCLS, 8, 3, Wgen_frag);
    swizzleB_kernel<<<(16 * 16 * 32 + 255) / 256, 256, 0, stream>>>(
        W_i2h, 512, 512, 256, 16, 16, Wi2h_frag);
    wg_swizzle_kernel<<<(24 * 64 * 32 + 255) / 256, 256, 0, stream>>>(
        W_ih, W_hh, Wg_frag);
    misc_prep_kernel<<<((NCLS + 1) * GATES + 255) / 256, 256, 0, stream>>>(
        W_ih, b_ih, b_hh, Wonehot, bcomb);

    hproj_kernel<<<(B_SZ * T_ATT) / 16, 256, 0, stream>>>(batch_H, Wi2h_frag, Hproj);

    decoder_kernel<<<B_SZ / 16, 256, 0, stream>>>(
        batch_H, text, Hproj, b_h2h, W_score,
        Wh2h_frag, Wg_frag, Wgen_frag, Wonehot, bcomb, b_gen, out);
}
